// table_splitautoencoder_template_77180562309401
// MI455X (gfx1250) — matrix operand order repaired and checked by reading, not compiled, not run on hardware
//
#include <hip/hip_runtime.h>
#include <hip/hip_bf16.h>

typedef __attribute__((ext_vector_type(16))) __bf16          v16bf;
typedef __attribute__((ext_vector_type(16))) unsigned short  v16us;
typedef __attribute__((ext_vector_type(8)))  float           v8f;

#define BATCH      16384
#define EBDIN      32
#define NVOCAB     4096
#define KPAD       64
#define NTILES     (NVOCAB / 16)   // 256
#define NPAIRS     (NTILES / 2)    // 128
#define NCHUNK     5
#define THREADS    (NCHUNK * 32)   // 160

__device__ __forceinline__ unsigned short f32_to_bf16_rne(float f) {
    unsigned u = __float_as_uint(f);
    u += 0x7FFFu + ((u >> 16) & 1u);
    return (unsigned short)(u >> 16);
}

// ---------------------------------------------------------------------------
// Pre-pass: H1 (45 x 4096 f32) -> H1t (4096 x 64 bf16), K-padded, with the
// per-column physical order [K0-15 | K32-47 | K16-31 | K48-63] so that the
// lane-half's two WMMA B operands are ONE contiguous 64-byte region.
// Logical k==45 carries the +64.0 argmax bias row (paired with A[k=45]=1.0),
// which makes every score positive at zero runtime cost.
// ---------------------------------------------------------------------------
__global__ void h1_to_bf16_t(const float* __restrict__ H1,
                             unsigned short* __restrict__ H1t) {
    int t  = blockIdx.x * blockDim.x + threadIdx.x;  // 0 .. 4096*64-1
    int n  = t >> 6;
    int k  = t & 63;                                  // logical K
    int kc = (k >> 5) & 1;
    int hh = (k >> 4) & 1;
    int r  = k & 15;
    int pos = hh * 32 + kc * 16 + r;                  // physical slot in column
    float v = (k < 45) ? H1[k * NVOCAB + n] : (k == 45 ? 64.0f : 0.0f);
    H1t[n * KPAD + pos] = f32_to_bf16_rne(v);
}

__device__ __forceinline__ v8f wmma_pair(const v16bf A0, const v16bf A1,
                                         v16us b0u, v16us b1u) {
    v16bf B0 = __builtin_bit_cast(v16bf, b0u);
    v16bf B1 = __builtin_bit_cast(v16bf, b1u);
    v8f a = {};
    a = __builtin_amdgcn_wmma_f32_16x16x32_bf16(false, A0, false, B0,
                                                (short)0, a, false, false);
    a = __builtin_amdgcn_wmma_f32_16x16x32_bf16(false, A1, false, B1,
                                                (short)0, a, false, false);
    return a;
}

// ---------------------------------------------------------------------------
// Main kernel: one block = one batch tile of 16 rows; 5 waves, one per chunk.
// npairs is a runtime argument so the scan loop cannot be fully unrolled.
// ---------------------------------------------------------------------------
__global__ void __launch_bounds__(THREADS)
autoenc_wmma_kernel(const int*   __restrict__ x,       // (B,1,2)
                    const float* __restrict__ lenLUT,  // (2048,32)
                    const float* __restrict__ ipdLUT,  // (2048,32)
                    const float* __restrict__ S1,      // (15,2,15)
                    const float* __restrict__ T1,      // (15,15)
                    const float* __restrict__ LUT1,    // (5,4096,32)
                    const float* __restrict__ S2,      // (1,2,15)
                    const float* __restrict__ H2,      // (15,16)
                    const float* __restrict__ T2,      // (1,15)
                    const float* __restrict__ LUT2,    // (1,16,32)
                    const unsigned short* __restrict__ H1t, // (4096,64) bf16
                    float* __restrict__ out,           // reconstruct | ebd
                    int npairs)
{
    __shared__ float sE[16 * EBDIN];
    __shared__ float sS1[15 * 2 * 15];
    __shared__ float sT1[15 * 15];
    __shared__ int   sIdx1[16 * NCHUNK];
    __shared__ int   sIdx2[16];

    const int tile = blockIdx.x;
    const int tid  = threadIdx.x;
    const int wave = tid >> 5;              // chunk index c
    const int lane = tid & 31;
    const int m    = lane & 15;             // A row, and B/C column
    const int h    = lane >> 4;             // lane half

    float* out_ebd = out + (size_t)BATCH * EBDIN;

    // ---- cooperative load phase -------------------------------------------
    for (int t = tid; t < 15 * 2 * 15; t += THREADS) sS1[t] = S1[t];
    for (int t = tid; t < 15 * 15;     t += THREADS) sT1[t] = T1[t];
    for (int t = tid; t < 16 * EBDIN;  t += THREADS) {
        int row = t >> 5, d = t & 31;
        int b   = tile * 16 + row;
        int i0  = x[b * 2 + 0];
        int i1  = x[b * 2 + 1];
        float e = lenLUT[i0 * EBDIN + d] + ipdLUT[i1 * EBDIN + d];
        sE[t] = e;
        out_ebd[(size_t)b * EBDIN + d] = e;   // second output: ebd
    }
    __syncthreads();

    // ---- tiny path: scores2 / idx2 (wave 0, one lane per row) -------------
    if (wave == 0 && lane < 16) {
        float e30 = sE[lane * EBDIN + 30];
        float e31 = sE[lane * EBDIN + 31];
        float bb[15];
        #pragma unroll
        for (int k = 0; k < 15; ++k) {
            float v = e30 * S2[k] + e31 * S2[15 + k] - T2[k];
            bb[k] = (v > 0.0f) ? 1.0f : -1.0f;   // where(v==0,-1,sign(v))
        }
        float best = -3.402823466e38f; int bi = 0;
        #pragma unroll
        for (int n = 0; n < 16; ++n) {
            float s = 0.0f;
            #pragma unroll
            for (int k = 0; k < 15; ++k) s += bb[k] * H2[k * 16 + n];
            if (s > best) { best = s; bi = n; }
        }
        sIdx2[lane] = bi;
    }

    // ---- build the 16x64 sign A-matrix (contiguous K per lane half) ------------
    // VGPR v: K = h*16 + v*2 (+kc*32); M = lane&15.
    // Logical k==45 is the bias lane (1.0), pairing with H1t's 64.0 row.
    const int c = wave;
    v16us a_us0, a_us1;
    #pragma unroll
    for (int kc = 0; kc < 2; ++kc) {
        #pragma unroll
        for (int v = 0; v < 8; ++v) {
            int kbase = kc * 32 + h * 16 + v * 2;
            #pragma unroll
            for (int p = 0; p < 2; ++p) {
                int k = kbase + p;
                unsigned short bits = 0;
                if (k < 45) {
                    int j  = c * 45 + k;
                    int cc = j / 15;
                    int kk = j - cc * 15;
                    float av = sE[m * EBDIN + 2 * cc]     * sS1[cc * 30 + kk]
                             + sE[m * EBDIN + 2 * cc + 1] * sS1[cc * 30 + 15 + kk]
                             - sT1[cc * 15 + kk] - 1.0e-4f;
                    bits = (av > 0.0f) ? (unsigned short)0x3F80
                         : (av < 0.0f) ? (unsigned short)0xBF80
                                       : (unsigned short)0x0000;
                } else if (k == 45) {
                    bits = (unsigned short)0x3F80;   // bias 1.0
                }
                if (kc == 0) a_us0[2 * v + p] = bits;
                else         a_us1[2 * v + p] = bits;
            }
        }
    }
    const v16bf A0 = __builtin_bit_cast(v16bf, a_us0);
    const v16bf A1 = __builtin_bit_cast(v16bf, a_us1);

    // ---- WMMA argmax sweep: 128 pairs of N-tiles, double pipeline ---------
    //  * loads for pair i+1 issued while computing pair i (no L2 stalls)
    //  * pair i-1 consumed while pair i's WMMAs are in flight (no hazard nops)
    //  * packed argmax: scores > 0 so IEEE bit order == unsigned order:
    //      packed = (bits & 0xFFFFF000) | (4095 - nidx);  running v_max_u32
    //  * dummy zero accumulators prime the pipeline (packed 0 never wins)
    unsigned maxp[8];
    #pragma unroll
    for (int r = 0; r < 8; ++r) maxp[r] = 0u;

    auto consume = [&](const v8f& a, int inv) {
        #pragma unroll
        for (int r = 0; r < 8; ++r) {
            unsigned pb = (__float_as_uint(a[r]) & 0xFFFFF000u) | (unsigned)inv;
            maxp[r] = (pb > maxp[r]) ? pb : maxp[r];
        }
    };

    const unsigned short* p = H1t + (size_t)m * KPAD + (size_t)h * 32;

    // preload pair 0
    v16us L0a = *(const v16us*)(p);
    v16us L1a = *(const v16us*)(p + 16);
    v16us L0b = *(const v16us*)(p + 1024);
    v16us L1b = *(const v16us*)(p + 1040);
    p += 2048;

    v8f accA = {}, accB = {};               // dummy pipeline stages
    int invA = 0, invB = 0;
    int invNext = 4095 - m;                 // tag of pair being computed

    #pragma unroll 2
    for (int i = 0; i < npairs - 1; ++i) {
        // prefetch pair i+1
        v16us N0a = *(const v16us*)(p);
        v16us N1a = *(const v16us*)(p + 16);
        v16us N0b = *(const v16us*)(p + 1024);
        v16us N1b = *(const v16us*)(p + 1040);
        p += 2048;

        // compute pair i, consuming pair i-1 in the shadow of the WMMAs
        v8f accA2 = wmma_pair(A0, A1, L0a, L1a);
        consume(accA, invA);
        v8f accB2 = wmma_pair(A0, A1, L0b, L1b);
        consume(accB, invB);

        accA = accA2; accB = accB2;
        invA = invNext; invB = invNext - 16;
        invNext -= 32;
        L0a = N0a; L1a = N1a; L0b = N0b; L1b = N1b;
    }

    // final pair: compute from preloaded regs, drain the pipeline
    {
        v8f accA2 = wmma_pair(A0, A1, L0a, L1a);
        consume(accA, invA);
        v8f accB2 = wmma_pair(A0, A1, L0b, L1b);
        consume(accB, invB);
        consume(accA2, invNext);
        consume(accB2, invNext - 16);
    }

    // ---- cross-lane argmax reduction within 16-lane halves ----------------
    #pragma unroll
    for (int off = 8; off; off >>= 1) {
        #pragma unroll
        for (int r = 0; r < 8; ++r) {
            unsigned op = (unsigned)__shfl_xor((int)maxp[r], off, 32);
            maxp[r] = (op > maxp[r]) ? op : maxp[r];
        }
    }
    if (m == 0) {
        #pragma unroll
        for (int r = 0; r < 8; ++r)
            sIdx1[(h * 8 + r) * NCHUNK + c] = 4095 - (int)(maxp[r] & 4095u);
    }
    __syncthreads();

    // ---- gather + sum: reconstruct = sum_c LUT1[c][idx1] + LUT2[idx2] -----
    for (int t = tid; t < 16 * EBDIN; t += THREADS) {
        int row = t >> 5, d = t & 31;
        float s = 0.0f;
        #pragma unroll
        for (int cc = 0; cc < NCHUNK; ++cc) {
            int idx = sIdx1[row * NCHUNK + cc];
            s += LUT1[((size_t)cc * NVOCAB + idx) * EBDIN + d];
        }
        s += LUT2[(size_t)sIdx2[row] * EBDIN + d];
        out[((size_t)tile * 16 + row) * EBDIN + d] = s;
    }
}

// ---------------------------------------------------------------------------
extern "C" void kernel_launch(void* const* d_in, const int* in_sizes, int n_in,
                              void* d_out, int out_size, void* d_ws, size_t ws_size,
                              hipStream_t stream) {
    const int*   x      = (const int*)  d_in[0];
    const float* lenLUT = (const float*)d_in[1];
    const float* ipdLUT = (const float*)d_in[2];
    const float* S1     = (const float*)d_in[3];
    const float* H1     = (const float*)d_in[4];
    const float* T1     = (const float*)d_in[5];
    const float* LUT1   = (const float*)d_in[6];
    const float* S2     = (const float*)d_in[7];
    const float* H2     = (const float*)d_in[8];
    const float* T2     = (const float*)d_in[9];
    const float* LUT2   = (const float*)d_in[10];
    float* out = (float*)d_out;

    unsigned short* H1t = (unsigned short*)d_ws;   // 4096*64*2 = 512 KB

    h1_to_bf16_t<<<(NVOCAB * KPAD) / 256, 256, 0, stream>>>(H1, H1t);

    autoenc_wmma_kernel<<<BATCH / 16, THREADS, 0, stream>>>(
        x, lenLUT, ipdLUT, S1, T1, LUT1, S2, H2, T2, LUT2, H1t, out, NPAIRS);

    (void)in_sizes; (void)n_in; (void)out_size; (void)ws_size;
}
